// MyViT_54528904790018
// MI455X (gfx1250) — matrix operand order repaired and checked by reading, not compiled, not run on hardware
//
#include <hip/hip_runtime.h>
#include <math.h>

// ---------------------------------------------------------------------------
// ViT-Base forward for MI455X (gfx1250), bf16 WMMA everywhere.
// Fragment loaders clamp (not branch+zero): out-of-range rows only feed
// masked-out C elements, so clamping keeps loads legal with zero divergence.
// ---------------------------------------------------------------------------

typedef __attribute__((ext_vector_type(16))) __bf16       v16bf;
typedef __attribute__((ext_vector_type(8)))  float        v8f;
typedef __attribute__((ext_vector_type(4)))  unsigned int v4u;

#define BN   32
#define CCH  3
#define NPP  14
#define PSZ  16
#define SS   197
#define SPAD 224          // S padded to multiple of 32 for WMMA K / alignment
#define NHH  12
#define DD   768
#define DHH  64
#define LL   12
#define FFF  3072
#define NOUT 1000
#define NPAT 196

// output modes for the generic GEMM epilogue
#define OUT_F32        0
#define OUT_BF16       1
#define OUT_GELU_BF16  2
#define OUT_ADD_F32    3
#define OUT_BF16_TRANS 4

__device__ __forceinline__ unsigned short f2bf(float x) {
  unsigned int u = __float_as_uint(x);
  unsigned int r = u + 0x7FFFu + ((u >> 16) & 1u);   // round-to-nearest-even
  return (unsigned short)(r >> 16);
}

union FragU { v4u q[2]; v16bf v; };

// A fragment: 16x32 bf16 (MxK). Per ISA: lane<16 rows M=lane, K={0..7}|{16..23};
// lane>=16 same rows, K={8..15}|{24..31}. Two 16B loads per lane, row clamped.
__device__ __forceinline__ v16bf load_a_frag(const unsigned short* __restrict__ base,
                                             int row0, long ld, int k, int maxRow, int lane) {
  int r = row0 + (lane & 15);
  r = (r < maxRow) ? r : (maxRow - 1);          // clamp: garbage rows are masked at store
  int ka = k + ((lane & 16) ? 8 : 0);
  const unsigned short* p = base + (long)r * ld + ka;
  FragU f;
  f.q[0] = *reinterpret_cast<const v4u*>(p);
  f.q[1] = *reinterpret_cast<const v4u*>(p + 16);
  return f.v;
}

// B fragment: 32x16 bf16 (KxN), sourced from row-major W[N,K] (torch convention,
// i.e. we load B^T rows). Per ISA B layout: lane%16 = n; lane<16 holds K={0..7}|{16..23},
// lane>=16 holds K={8..15}|{24..31} (matches A). Two 16B loads per lane, row clamped.
__device__ __forceinline__ v16bf load_b_frag(const unsigned short* __restrict__ base,
                                             int row0, long ld, int k, int maxRow, int lane) {
  int r = row0 + (lane & 15);
  r = (r < maxRow) ? r : (maxRow - 1);          // clamp: garbage cols are masked at store
  int kb = k + ((lane & 16) ? 8 : 0);
  const unsigned short* p = base + (long)r * ld + kb;
  FragU f;
  f.q[0] = *reinterpret_cast<const v4u*>(p);
  f.q[1] = *reinterpret_cast<const v4u*>(p + 16);
  return f.v;
}

// Generic batched GEMM: C[M,N] = epilogue(A[M,K] @ B^T + bias)
//   A  : bf16 row-major, leading dim lda, batch offsets sA0*z0 + sA1*z1
//   Bm : bf16 row-major [N,K] (weights), ldb, batch offsets sB0*z0 + sB1*z1
//   C  : fp32 or bf16 depending on mode
// One wave (32 threads) per 16(M) x 64(N) tile; A fragment reused 4x.
__global__ __launch_bounds__(32) void gemm_wmma_kernel(
    const unsigned short* __restrict__ A,  long lda, long sA0, long sA1,
    const unsigned short* __restrict__ Bm, long ldb, long sB0, long sB1,
    const float* __restrict__ bias, long sb0, long sb1,
    void* __restrict__ Cp, long ldc, long sC0, long sC1,
    int M, int N, int K, int nz0, int mode)
{
  const int lane = threadIdx.x & 31;
  const int z  = blockIdx.z;
  const int z0 = z % nz0;
  const int z1 = z / nz0;

  const unsigned short* Ab = A  + (long)sA0 * z0 + (long)sA1 * z1;
  const unsigned short* Bb = Bm + (long)sB0 * z0 + (long)sB1 * z1;
  const float* biasb = bias ? (bias + sb0 * z0 + sb1 * z1) : (const float*)0;
  const bool cF32 = (mode == OUT_F32 || mode == OUT_ADD_F32);
  char* Cb = (char*)Cp + ((long)sC0 * z0 + (long)sC1 * z1) * (cF32 ? 4 : 2);

  const int m0 = blockIdx.x * 16;
  const int n0 = blockIdx.y * 64;

  v8f acc[4] = {};
  for (int k = 0; k < K; k += 32) {
    // prefetch A two K-blocks ahead (global_prefetch_b8 on gfx1250)
    __builtin_prefetch(Ab + (long)(m0 + (lane & 15)) * lda + k + 64, 0, 1);

    v16bf a = load_a_frag(Ab, m0, lda, k, M, lane);
#pragma unroll
    for (int j = 0; j < 4; ++j) {
      v16bf b = load_b_frag(Bb, n0 + 16 * j, ldb, k, N, lane);
      acc[j] = __builtin_amdgcn_wmma_f32_16x16x32_bf16(
          /*neg_a=*/false, a, /*neg_b=*/false, b,
          /*c_mod=*/(short)0, acc[j], /*reuse_a=*/false, /*reuse_b=*/false);
    }
  }

  // C/D layout: VGPR v holds M = m0 + v (+8 for lanes>=16), N = n0+16j+(lane&15)
  const int rb = m0 + ((lane & 16) ? 8 : 0);
  const int cn = lane & 15;
#pragma unroll
  for (int j = 0; j < 4; ++j) {
    int col = n0 + 16 * j + cn;
    if (col >= N) continue;
    float bvv = biasb ? biasb[col] : 0.0f;
#pragma unroll
    for (int v = 0; v < 8; ++v) {
      int row = rb + v;
      if (row >= M) continue;
      float val = acc[j][v] + bvv;
      if (mode == OUT_GELU_BF16)
        val = 0.5f * val * (1.0f + erff(val * 0.70710678118654752f));
      if (mode == OUT_F32) {
        reinterpret_cast<float*>(Cb)[(long)row * ldc + col] = val;
      } else if (mode == OUT_ADD_F32) {
        float* c = reinterpret_cast<float*>(Cb) + (long)row * ldc + col;
        *c = *c + val;
      } else if (mode == OUT_BF16_TRANS) {
        reinterpret_cast<unsigned short*>(Cb)[(long)col * ldc + row] = f2bf(val);
      } else { // OUT_BF16 / OUT_GELU_BF16
        reinterpret_cast<unsigned short*>(Cb)[(long)row * ldc + col] = f2bf(val);
      }
    }
  }
}

// ---------------------------------------------------------------------------
// Support kernels
// ---------------------------------------------------------------------------

__global__ void f32_to_bf16_kernel(const float* __restrict__ in,
                                   unsigned short* __restrict__ out, long n) {
  long i = (long)blockIdx.x * blockDim.x + threadIdx.x;
  long stride = (long)gridDim.x * blockDim.x;
  for (; i < n; i += stride) out[i] = f2bf(in[i]);
}

// images [B,3,224,224] fp32 -> patches bf16 [B*196, 768], channel-major per patch
__global__ void patchify_kernel(const float* __restrict__ img,
                                unsigned short* __restrict__ out) {
  long idx = (long)blockIdx.x * 256 + threadIdx.x;
  if (idx >= (long)BN * NPAT * DD) return;
  int q = (int)(idx % DD);
  long t = idx / DD;
  int p = (int)(t % NPAT);
  int b = (int)(t / NPAT);
  int c  = q / (PSZ * PSZ);
  int r  = q % (PSZ * PSZ);
  int py = r / PSZ, px = r % PSZ;
  int pr = p / NPP, pc = p % NPP;
  long src = (((long)(b * CCH + c) * (NPP * PSZ)) + pr * PSZ + py) * (NPP * PSZ)
             + pc * PSZ + px;
  out[idx] = f2bf(img[src]);
}

// X[b,0,:] = cls + pos[0]; X[b,s>0,:] += pos[s]   (pos computed on the fly)
__global__ void cls_pos_kernel(float* __restrict__ X, const float* __restrict__ cls) {
  long idx = (long)blockIdx.x * 256 + threadIdx.x;
  if (idx >= (long)BN * SS * DD) return;
  int d = (int)(idx % DD);
  long t = idx / DD;
  int s = (int)(t % SS);
  float ang = (float)s / powf(10000.0f, 2.0f * (float)d / (float)DD);
  float pe  = ((d & 1) == 0) ? sinf(ang) : cosf(ang);
  float base = (s == 0) ? cls[d] : X[idx];
  X[idx] = base + pe;
}

// LayerNorm over D=768, fp32 in -> bf16 out. One 256-thread block per row.
__global__ __launch_bounds__(256) void layernorm_kernel(
    const float* __restrict__ X, const float* __restrict__ g,
    const float* __restrict__ bta, unsigned short* __restrict__ out) {
  long row = blockIdx.x;
  const float* x = X + row * DD;
  unsigned short* o = out + row * DD;
  __shared__ float red[256];
  float s = 0.0f;
  for (int i = threadIdx.x; i < DD; i += 256) s += x[i];
  red[threadIdx.x] = s; __syncthreads();
  for (int st = 128; st; st >>= 1) {
    if (threadIdx.x < st) red[threadIdx.x] += red[threadIdx.x + st];
    __syncthreads();
  }
  float mean = red[0] * (1.0f / DD);
  __syncthreads();
  float vs = 0.0f;
  for (int i = threadIdx.x; i < DD; i += 256) { float d = x[i] - mean; vs += d * d; }
  red[threadIdx.x] = vs; __syncthreads();
  for (int st = 128; st; st >>= 1) {
    if (threadIdx.x < st) red[threadIdx.x] += red[threadIdx.x + st];
    __syncthreads();
  }
  float inv = rsqrtf(red[0] * (1.0f / DD) + 1e-5f);
  for (int i = threadIdx.x; i < DD; i += 256)
    o[i] = f2bf((x[i] - mean) * inv * g[i] + bta[i]);
}

// softmax over attention rows: fp32 scores [.., SPAD] -> bf16 probs (pad = 0)
__global__ __launch_bounds__(256) void softmax_att_kernel(
    const float* __restrict__ att, unsigned short* __restrict__ probs, float scale) {
  long row = blockIdx.x;            // B*NH*S rows
  const float* a = att + row * SPAD;
  unsigned short* p = probs + row * SPAD;
  int t = threadIdx.x;
  __shared__ float red[256];
  float v = (t < SS) ? a[t] * scale : -3.4e38f;
  red[t] = v; __syncthreads();
  for (int st = 128; st; st >>= 1) {
    if (t < st) red[t] = fmaxf(red[t], red[t + st]);
    __syncthreads();
  }
  float mx = red[0]; __syncthreads();
  float e = (t < SS) ? __expf(v - mx) : 0.0f;
  red[t] = e; __syncthreads();
  for (int st = 128; st; st >>= 1) {
    if (t < st) red[t] += red[t + st];
    __syncthreads();
  }
  float inv = 1.0f / red[0];
  if (t < SPAD) p[t] = f2bf((t < SS) ? e * inv : 0.0f);
}

__global__ void extract_cls_kernel(const float* __restrict__ X,
                                   unsigned short* __restrict__ out) {
  int idx = blockIdx.x * 256 + threadIdx.x;
  if (idx >= BN * DD) return;
  int b = idx / DD, d = idx % DD;
  out[idx] = f2bf(X[(long)b * SS * DD + d]);
}

// generic row softmax fp32->fp32 (final head)
__global__ __launch_bounds__(256) void softmax_row_kernel(
    const float* __restrict__ in, float* __restrict__ out, int n) {
  int row = blockIdx.x;
  const float* x = in + (long)row * n;
  float* y = out + (long)row * n;
  __shared__ float red[256];
  float mx = -3.4e38f;
  for (int i = threadIdx.x; i < n; i += 256) mx = fmaxf(mx, x[i]);
  red[threadIdx.x] = mx; __syncthreads();
  for (int st = 128; st; st >>= 1) {
    if (threadIdx.x < st) red[threadIdx.x] = fmaxf(red[threadIdx.x], red[threadIdx.x + st]);
    __syncthreads();
  }
  mx = red[0]; __syncthreads();
  float s = 0.0f;
  for (int i = threadIdx.x; i < n; i += 256) s += __expf(x[i] - mx);
  red[threadIdx.x] = s; __syncthreads();
  for (int st = 128; st; st >>= 1) {
    if (threadIdx.x < st) red[threadIdx.x] += red[threadIdx.x + st];
    __syncthreads();
  }
  float inv = 1.0f / red[0];
  for (int i = threadIdx.x; i < n; i += 256) y[i] = __expf(x[i] - mx) * inv;
}

// ---------------------------------------------------------------------------
// Host orchestration
// ---------------------------------------------------------------------------

extern "C" void kernel_launch(void* const* d_in, const int* in_sizes, int n_in,
                              void* d_out, int out_size, void* d_ws, size_t ws_size,
                              hipStream_t stream) {
  (void)in_sizes; (void)n_in; (void)out_size; (void)ws_size;
  const float* images = (const float*)d_in[0];
  const float* W_map  = (const float*)d_in[1];
  const float* b_map  = (const float*)d_in[2];
  const float* cls    = (const float*)d_in[3];
  const float* ln1_g  = (const float*)d_in[4];
  const float* ln1_b  = (const float*)d_in[5];
  const float* Wq     = (const float*)d_in[6];
  const float* Wk     = (const float*)d_in[7];
  const float* Wv     = (const float*)d_in[8];
  const float* bq     = (const float*)d_in[9];
  const float* bk     = (const float*)d_in[10];
  const float* bv     = (const float*)d_in[11];
  const float* ln2_g  = (const float*)d_in[12];
  const float* ln2_b  = (const float*)d_in[13];
  const float* W1     = (const float*)d_in[14];
  const float* b1     = (const float*)d_in[15];
  const float* W2     = (const float*)d_in[16];
  const float* b2     = (const float*)d_in[17];
  const float* W_out  = (const float*)d_in[18];
  const float* b_out  = (const float*)d_in[19];

  char* ws = (char*)d_ws;
  size_t off = 0;
  auto alloc = [&](size_t bytes) -> char* {
    char* p = ws + off;
    off += (bytes + 255) & ~(size_t)255;
    return p;
  };

  unsigned short* wmap_b = (unsigned short*)alloc((size_t)DD * DD * 2);
  unsigned short* wq_b   = (unsigned short*)alloc((size_t)LL * NHH * DHH * DHH * 2);
  unsigned short* wk_b   = (unsigned short*)alloc((size_t)LL * NHH * DHH * DHH * 2);
  unsigned short* wv_b   = (unsigned short*)alloc((size_t)LL * NHH * DHH * DHH * 2);
  unsigned short* w1_b   = (unsigned short*)alloc((size_t)LL * FFF * DD * 2);
  unsigned short* w2_b   = (unsigned short*)alloc((size_t)LL * DD * FFF * 2);
  unsigned short* wout_b = (unsigned short*)alloc((size_t)NOUT * DD * 2);
  unsigned short* Pm     = (unsigned short*)alloc((size_t)BN * NPAT * DD * 2);
  float*          X      = (float*)         alloc((size_t)BN * SS * DD * 4);
  unsigned short* Hb     = (unsigned short*)alloc((size_t)BN * SS * DD * 2);
  unsigned short* Qb     = (unsigned short*)alloc((size_t)BN * SS * DD * 2);
  unsigned short* Kb     = (unsigned short*)alloc((size_t)BN * SS * DD * 2);
  unsigned short* Vt     = (unsigned short*)alloc((size_t)BN * NHH * DHH * SPAD * 2);
  float*          att    = (float*)         alloc((size_t)BN * NHH * SS * SPAD * 4);
  unsigned short* probs  = (unsigned short*)alloc((size_t)BN * NHH * SS * SPAD * 2);
  unsigned short* ffn1   = (unsigned short*)alloc((size_t)BN * SS * FFF * 2);
  unsigned short* xcls   = (unsigned short*)alloc((size_t)BN * DD * 2);
  float*          logits = (float*)         alloc((size_t)BN * NOUT * 4);

  auto conv = [&](const float* src, unsigned short* dst, long n) {
    f32_to_bf16_kernel<<<dim3(2048), dim3(256), 0, stream>>>(src, dst, n);
  };
  auto gemm = [&](const unsigned short* A, long lda, long sA0, long sA1,
                  const unsigned short* Bm, long ldb, long sB0, long sB1,
                  const float* bias, long sb0, long sb1,
                  void* C, long ldc, long sC0, long sC1,
                  int M, int N, int K, int nz0, int Z, int mode) {
    dim3 grid((unsigned)((M + 15) / 16), (unsigned)((N + 63) / 64), (unsigned)Z);
    gemm_wmma_kernel<<<grid, dim3(32), 0, stream>>>(
        A, lda, sA0, sA1, Bm, ldb, sB0, sB1, bias, sb0, sb1,
        C, ldc, sC0, sC1, M, N, K, nz0, mode);
  };

  // ---- weight conversion fp32 -> bf16 (re-done every call; deterministic) ----
  conv(W_map, wmap_b, (long)DD * DD);
  conv(Wq, wq_b, (long)LL * NHH * DHH * DHH);
  conv(Wk, wk_b, (long)LL * NHH * DHH * DHH);
  conv(Wv, wv_b, (long)LL * NHH * DHH * DHH);
  conv(W1, w1_b, (long)LL * FFF * DD);
  conv(W2, w2_b, (long)LL * DD * FFF);
  conv(W_out, wout_b, (long)NOUT * DD);

  // ---- patchify + patch embedding into X rows 1..196 per image ----
  {
    long n = (long)BN * NPAT * DD;
    patchify_kernel<<<dim3((unsigned)((n + 255) / 256)), dim3(256), 0, stream>>>(images, Pm);
  }
  const long BSD = (long)SS * DD;
  gemm(Pm, DD, 0, (long)NPAT * DD,
       wmap_b, DD, 0, 0,
       b_map, 0, 0,
       X + DD /* row 1 of each image */, DD, 0, BSD,
       NPAT, DD, DD, 1, BN, OUT_F32);
  {
    long n = (long)BN * SS * DD;
    cls_pos_kernel<<<dim3((unsigned)((n + 255) / 256)), dim3(256), 0, stream>>>(X, cls);
  }

  // ---- transformer layers ----
  for (int l = 0; l < LL; ++l) {
    layernorm_kernel<<<dim3(BN * SS), dim3(256), 0, stream>>>(
        X, ln1_g + (long)l * DD, ln1_b + (long)l * DD, Hb);

    // per-head Q/K projections (Z = B*NH; z0 = head, z1 = batch)
    gemm(Hb, DD, DHH, BSD,
         wq_b + (long)l * NHH * DHH * DHH, DHH, (long)DHH * DHH, 0,
         bq + (long)l * NHH * DHH, DHH, 0,
         Qb, DD, DHH, BSD,
         SS, DHH, DHH, NHH, BN * NHH, OUT_BF16);
    gemm(Hb, DD, DHH, BSD,
         wk_b + (long)l * NHH * DHH * DHH, DHH, (long)DHH * DHH, 0,
         bk + (long)l * NHH * DHH, DHH, 0,
         Kb, DD, DHH, BSD,
         SS, DHH, DHH, NHH, BN * NHH, OUT_BF16);
    // V projection, stored transposed: Vt[b,h,e,t] (ld = SPAD)
    gemm(Hb, DD, DHH, BSD,
         wv_b + (long)l * NHH * DHH * DHH, DHH, (long)DHH * DHH, 0,
         bv + (long)l * NHH * DHH, DHH, 0,
         Vt, SPAD, (long)DHH * SPAD, (long)NHH * DHH * SPAD,
         SS, DHH, DHH, NHH, BN * NHH, OUT_BF16_TRANS);

    // scores[s,t] = Q[s,:] . K[t,:]  (fp32, ld SPAD)
    gemm(Qb, DD, DHH, BSD,
         Kb, DD, DHH, BSD,
         nullptr, 0, 0,
         att, SPAD, (long)SS * SPAD, (long)NHH * SS * SPAD,
         SS, SS, DHH, NHH, BN * NHH, OUT_F32);

    softmax_att_kernel<<<dim3(BN * NHH * SS), dim3(256), 0, stream>>>(
        att, probs, 0.125f /* 1/sqrt(64) */);

    // attn-out = probs @ V, accumulated into residual X
    gemm(probs, SPAD, (long)SS * SPAD, (long)NHH * SS * SPAD,
         Vt, SPAD, (long)DHH * SPAD, (long)NHH * DHH * SPAD,
         nullptr, 0, 0,
         X, DD, DHH, BSD,
         SS, DHH, SPAD, NHH, BN * NHH, OUT_ADD_F32);

    layernorm_kernel<<<dim3(BN * SS), dim3(256), 0, stream>>>(
        X, ln2_g + (long)l * DD, ln2_b + (long)l * DD, Hb);

    // FFN1: [B*S, 768] -> GELU -> bf16 [B*S, 3072]
    gemm(Hb, DD, 0, 0,
         w1_b + (long)l * FFF * DD, DD, 0, 0,
         b1 + (long)l * FFF, 0, 0,
         ffn1, FFF, 0, 0,
         BN * SS, FFF, DD, 1, 1, OUT_GELU_BF16);
    // FFN2: [B*S, 3072] -> += X
    gemm(ffn1, FFF, 0, 0,
         w2_b + (long)l * DD * FFF, FFF, 0, 0,
         b2 + (long)l * DD, 0, 0,
         X, DD, 0, 0,
         BN * SS, DD, FFF, 1, 1, OUT_ADD_F32);
  }

  // ---- classification head on cls token ----
  extract_cls_kernel<<<dim3((BN * DD + 255) / 256), dim3(256), 0, stream>>>(X, xcls);
  gemm(xcls, DD, 0, 0,
       wout_b, DD, 0, 0,
       b_out, 0, 0,
       logits, NOUT, 0, 0,
       BN, NOUT, DD, 1, 1, OUT_F32);
  softmax_row_kernel<<<dim3(BN), dim3(256), 0, stream>>>(logits, (float*)d_out, NOUT);
}
